// CrossAttention1D_20873541058743
// MI455X (gfx1250) — matrix operand order repaired and checked by reading, not compiled, not run on hardware
//
#include <hip/hip_runtime.h>
#include <hip/hip_bf16.h>

// ---------------------------------------------------------------------------
// MI455X (gfx1250, wave32) cross-attention, bf16 WMMA + async-to-LDS staging.
// ---------------------------------------------------------------------------

typedef __attribute__((ext_vector_type(16))) __bf16 v16bf;
typedef __attribute__((ext_vector_type(8)))  __bf16 bf16x8;
typedef __attribute__((ext_vector_type(8)))  float  v8f;
typedef __attribute__((ext_vector_type(4)))  int    v4i;

union Frag { v16bf v; bf16x8 h[2]; };

#define WMMA_BF16(A, B, C) \
  __builtin_amdgcn_wmma_f32_16x16x32_bf16(false, (A), false, (B), (short)0, (C), false, false)

typedef __attribute__((address_space(1))) v4i* gptr_v4i;   // global int4*
typedef __attribute__((address_space(3))) v4i* lptr_v4i;   // LDS int4*

// 16-byte global -> LDS async copy (ASYNCcnt-tracked, no VGPR round trip).
__device__ __forceinline__ void async_copy16(const __bf16* g, __bf16* l) {
#if __has_builtin(__builtin_amdgcn_global_load_async_to_lds_b128)
  __builtin_amdgcn_global_load_async_to_lds_b128(
      (gptr_v4i)(uintptr_t)g, (lptr_v4i)(unsigned)(uintptr_t)l, 0, 0);
#else
  asm volatile("global_load_async_to_lds_b128 %0, %1, off"
               :: "v"((unsigned)(uintptr_t)l), "v"(g) : "memory");
#endif
}

__device__ __forceinline__ void wait_async0() {
#if __has_builtin(__builtin_amdgcn_s_wait_asynccnt)
  __builtin_amdgcn_s_wait_asynccnt(0);
#else
  asm volatile("s_wait_asynccnt 0" ::: "memory");
#endif
}

__device__ __forceinline__ float halfwave_max(float x) {
  // lanes 0-15 and 16-31 reduce independently (xor masks < 16 stay in-half)
  for (int m = 1; m < 16; m <<= 1) x = fmaxf(x, __shfl_xor(x, m, 32));
  return x;
}
__device__ __forceinline__ float halfwave_sum(float x) {
  for (int m = 1; m < 16; m <<= 1) x += __shfl_xor(x, m, 32);
  return x;
}

// ------------------------------ elementwise --------------------------------

__global__ void xattn_cvt_f32_bf16(const float* __restrict__ src,
                                   __bf16* __restrict__ dst, int n) {
  int i = blockIdx.x * blockDim.x + threadIdx.x;
  if (i < n) dst[i] = (__bf16)src[i];
}

__global__ void xattn_zero_f32(float* __restrict__ p, int n) {
  int i = blockIdx.x * blockDim.x + threadIdx.x;
  if (i < n) p[i] = 0.0f;
}

// ------------------------------ GEMM + bias --------------------------------
// C[8192,1024] = A[8192,1024] @ W[1024,1024] + bias. Block tile 128x128,
// 8 waves (2x4), wave tile 64x32 (4x2 accumulators), K-step 32.
// MODE 0: bf16 permuted [B=8, H=16, n, d] (q/k/v proj)
// MODE 1: f32 row-major [8192,1024]       (final proj)

template <int MODE>
__global__ __launch_bounds__(256) void xattn_gemm_bf16_wmma(
    const __bf16* __restrict__ A, const __bf16* __restrict__ W,
    const float* __restrict__ bias, __bf16* __restrict__ outb,
    float* __restrict__ outf) {
  __shared__ __bf16 As[128][32];   // A tile, row-major (async-filled)
  __shared__ __bf16 Wt[128][32];   // W tile TRANSPOSED: Wt[n][k]

  const int t    = threadIdx.x;
  const int m0   = blockIdx.y * 128;
  const int n0   = blockIdx.x * 128;
  const int wave = t >> 5, lane = t & 31;
  const int half = lane >> 4, m = lane & 15;
  const int wm   = wave >> 2;      // 0..1 -> 64 rows each
  const int wn   = wave & 3;       // 0..3 -> 32 cols each

  const int arow = t >> 1, ahh = (t & 1) * 16;     // A staging coords
  const int wkk  = t >> 3, wnc = (t & 7) * 16;     // W staging coords

  v8f acc[4][2];
  const v8f vzero = {0.f, 0.f, 0.f, 0.f, 0.f, 0.f, 0.f, 0.f};
  for (int i = 0; i < 4; i++)
    for (int j = 0; j < 2; j++) acc[i][j] = vzero;

  for (int k0 = 0; k0 < 1024; k0 += 32) {
    __syncthreads();   // previous iteration's fragment reads done
    {  // stage A via async global->LDS (2x16B per thread)
      const __bf16* s = A + (((size_t)(m0 + arow)) << 10) + k0 + ahh;
      async_copy16(s,     &As[arow][ahh]);
      async_copy16(s + 8, &As[arow][ahh + 8]);
    }
    {  // stage W transposed: coalesced global read along n, scatter to Wt[n][k]
      const __bf16* s = W + (((size_t)(k0 + wkk)) << 10) + n0 + wnc;
      bf16x8 w0 = ((const bf16x8*)s)[0], w1 = ((const bf16x8*)s)[1];
      for (int e = 0; e < 8; e++) {
        Wt[wnc + e][wkk]     = w0[e];
        Wt[wnc + 8 + e][wkk] = w1[e];
      }
    }
    if (k0 + 32 < 1024) {  // prefetch next K-step lines into cache
      __builtin_prefetch(A + (((size_t)(m0 + arow)) << 10) + k0 + 32 + ahh, 0, 0);
      __builtin_prefetch(W + (((size_t)(k0 + 32 + wkk)) << 10) + n0 + wnc, 0, 0);
    }
    wait_async0();
    __syncthreads();

    // A fragment (16x32): lanes 0-15 hold K0-7 & K16-23 of row m, lanes 16-31
    // hold K8-15 & K24-31 (ISA 7.12.2 16-bit A layout).
    Frag aF[4], bF[2];
    for (int i = 0; i < 4; i++) {
      const __bf16* s = &As[wm * 64 + i * 16 + m][0];
      aF[i].h[0] = *(const bf16x8*)(s + half * 8);
      aF[i].h[1] = *(const bf16x8*)(s + 16 + half * 8);
    }
    // B fragment (32x16): column n = lane&15; lanes 0-15 hold K0-7 & K16-23,
    // lanes 16-31 hold K8-15 & K24-31 (same k order as A) from transposed tile.
    for (int j = 0; j < 2; j++) {
      const __bf16* s = &Wt[wn * 32 + j * 16 + m][0];
      bF[j].h[0] = *(const bf16x8*)(s + half * 8);
      bF[j].h[1] = *(const bf16x8*)(s + 16 + half * 8);
    }
    for (int i = 0; i < 4; i++)
      for (int j = 0; j < 2; j++)
        acc[i][j] = WMMA_BF16(aF[i].v, bF[j].v, acc[i][j]);
  }

  // Epilogue. C/D layout: VGPR v <-> row v+8*half, col = lane&15.
  if (MODE == 0) {
    const int bb = m0 >> 10;                       // batch (128-row blocks never straddle)
    for (int i = 0; i < 4; i++) {
      const int nr = (m0 & 1023) + wm * 64 + i * 16 + 8 * half;
      for (int j = 0; j < 2; j++) {
        const int col = n0 + wn * 32 + j * 16 + m;
        const int hh = col >> 6, dd = col & 63;
        const float bvv = bias[col];
        __bf16* dst = outb + (((size_t)bb << 20) + ((size_t)hh << 16) +
                              ((size_t)nr << 6) + dd);
        for (int vv = 0; vv < 8; vv++)
          dst[(size_t)vv << 6] = (__bf16)(acc[i][j][vv] + bvv);
      }
    }
  } else {
    for (int i = 0; i < 4; i++) {
      const int row = m0 + wm * 64 + i * 16 + 8 * half;
      for (int j = 0; j < 2; j++) {
        const int col = n0 + wn * 32 + j * 16 + m;
        const float bvv = bias[col];
        float* dst = outf + (((size_t)row << 10) + col);
        for (int vv = 0; vv < 8; vv++)
          dst[(size_t)vv << 10] = acc[i][j][vv] + bvv;
      }
    }
  }
}

// ------------------------------ attention ----------------------------------
// q,k,v: bf16 [B,H,N,D] (B=8,H=16,N=1024,D=64). One wave per 16-row q tile,
// flash-style online softmax over 32-key chunks. Raw scaled scores go to
// attn_avg via atomicAdd (zeroed each launch). Output written bf16 [B,N,C].

__global__ __launch_bounds__(256) void xattn_flash_wmma(
    const __bf16* __restrict__ qm, const __bf16* __restrict__ km,
    const __bf16* __restrict__ vm, __bf16* __restrict__ om,
    float* __restrict__ aavg) {
  __shared__ __bf16 Pl[8][16][32];   // per-wave P tile (q x 32 keys)
  __shared__ __bf16 Vl[8][64][32];   // per-wave V chunk transposed [d][key]

  const float SCALE = 0.125f;        // 64^-0.5
  const int t    = threadIdx.x;
  const int wave = t >> 5, lane = t & 31;
  const int half = lane >> 4, m = lane & 15;
  const int bh = blockIdx.x;         // 0..127
  const int b  = bh >> 4, h = bh & 15;
  const int q0 = blockIdx.y * 128 + wave * 16;

  const __bf16* Qb = qm + ((size_t)bh << 16);
  const __bf16* Kb = km + ((size_t)bh << 16);
  const __bf16* Vb = vm + ((size_t)bh << 16);

  // Q fragments for d=[0,32) and d=[32,64), reused across all key chunks.
  Frag aQ[2];
  for (int dh = 0; dh < 2; dh++) {
    const __bf16* s = Qb + (((size_t)(q0 + m)) << 6) + dh * 32;
    aQ[dh].h[0] = *(const bf16x8*)(s + half * 8);
    aQ[dh].h[1] = *(const bf16x8*)(s + 16 + half * 8);
  }

  float mstat[8], lstat[8];
  for (int i = 0; i < 8; i++) { mstat[i] = -1e30f; lstat[i] = 0.0f; }
  const v8f vzero = {0.f, 0.f, 0.f, 0.f, 0.f, 0.f, 0.f, 0.f};
  v8f oacc[4];
  for (int dt = 0; dt < 4; dt++) oacc[dt] = vzero;

  // per-lane attn_avg base: row block (q0 + 8*half), col = j0 + m
  float* arow = aavg + ((size_t)b << 20) + (((size_t)(q0 + 8 * half)) << 10) + m;

  for (int j0 = 0; j0 < 1024; j0 += 32) {
    // B fragments of K^T: T[n][kk] = K[j0+sub*16+n][dh*32+kk] (contiguous rows)
    Frag bK[2][2];
    for (int kt = 0; kt < 2; kt++)
      for (int dh = 0; dh < 2; dh++) {
        const __bf16* s = Kb + (((size_t)(j0 + kt * 16 + m)) << 6) + dh * 32;
        bK[kt][dh].h[0] = *(const bf16x8*)(s + half * 8);
        bK[kt][dh].h[1] = *(const bf16x8*)(s + 16 + half * 8);
      }
    if (j0 + 32 < 1024) {  // prefetch next chunk's K/V rows
      __builtin_prefetch(Kb + (((size_t)(j0 + 32 + lane)) << 6), 0, 0);
      __builtin_prefetch(Vb + (((size_t)(j0 + 32 + lane)) << 6), 0, 0);
    }
    v8f s0 = vzero, s1 = vzero;
    s0 = WMMA_BF16(aQ[0].v, bK[0][0].v, s0);
    s0 = WMMA_BF16(aQ[1].v, bK[0][1].v, s0);
    s1 = WMMA_BF16(aQ[0].v, bK[1][0].v, s1);
    s1 = WMMA_BF16(aQ[1].v, bK[1][1].v, s1);

    // scale, accumulate attn_avg (sum over heads / H), online softmax
    float p0[8], p1[8], corr[8];
    for (int vv = 0; vv < 8; vv++) {
      float a0 = (float)s0[vv] * SCALE;
      float a1 = (float)s1[vv] * SCALE;
      atomicAdd(arow + ((size_t)vv << 10) + j0,      a0 * 0.0625f);
      atomicAdd(arow + ((size_t)vv << 10) + j0 + 16, a1 * 0.0625f);
      p0[vv] = a0; p1[vv] = a1;
    }
    for (int vv = 0; vv < 8; vv++) {
      float cm   = halfwave_max(fmaxf(p0[vv], p1[vv]));
      float mnew = fmaxf(mstat[vv], cm);
      corr[vv]   = __expf(mstat[vv] - mnew);
      mstat[vv]  = mnew;
      p0[vv]     = __expf(p0[vv] - mnew);
      p1[vv]     = __expf(p1[vv] - mnew);
      lstat[vv]  = lstat[vv] * corr[vv] + halfwave_sum(p0[vv] + p1[vv]);
    }

    __syncthreads();   // prior chunk's LDS reads complete block-wide
    // P tile -> LDS, row-major [q][key] so the A fragment reads are contiguous
    for (int vv = 0; vv < 8; vv++) {
      int rl = vv + 8 * half;
      Pl[wave][rl][m]      = (__bf16)p0[vv];
      Pl[wave][rl][16 + m] = (__bf16)p1[vv];
    }
    // V chunk 32x64 -> LDS transposed [d][key]: lane owns one key row
    {
      const __bf16* vs = Vb + (((size_t)(j0 + lane)) << 6);
      for (int i = 0; i < 8; i++) {
        bf16x8 vv8 = *(const bf16x8*)(vs + i * 8);
        for (int e = 0; e < 8; e++) Vl[wave][i * 8 + e][lane] = vv8[e];
      }
    }
    __syncthreads();

    for (int dt = 0; dt < 4; dt++)
      for (int vv = 0; vv < 8; vv++) oacc[dt][vv] *= corr[vv];

    Frag aP;
    aP.h[0] = *(const bf16x8*)&Pl[wave][m][half * 8];
    aP.h[1] = *(const bf16x8*)&Pl[wave][m][16 + half * 8];
    for (int dt = 0; dt < 4; dt++) {
      Frag bV;   // T[n][kk] = V[j0+kk][dt*16+n] = Vl[dt*16+n][kk]
      bV.h[0] = *(const bf16x8*)&Vl[wave][dt * 16 + m][half * 8];
      bV.h[1] = *(const bf16x8*)&Vl[wave][dt * 16 + m][16 + half * 8];
      oacc[dt] = WMMA_BF16(aP.v, bV.v, oacc[dt]);
    }
  }

  // normalize and write [B, N, C] bf16 (feeds the output projection GEMM)
  __bf16* obase = om + ((size_t)b << 20) + (((size_t)(q0 + 8 * half)) << 10) +
                  (h << 6) + m;
  for (int dt = 0; dt < 4; dt++)
    for (int vv = 0; vv < 8; vv++) {
      float val = oacc[dt][vv] / lstat[vv];
      obase[((size_t)vv << 10) + dt * 16] = (__bf16)val;
    }
}

// ------------------------------ launcher -----------------------------------

extern "C" void kernel_launch(void* const* d_in, const int* in_sizes, int n_in,
                              void* d_out, int out_size, void* d_ws, size_t ws_size,
                              hipStream_t stream) {
  (void)in_sizes; (void)n_in; (void)out_size; (void)ws_size;
  const float* xq = (const float*)d_in[0];
  const float* xk = (const float*)d_in[1];
  const float* xv = (const float*)d_in[2];
  const float* Wq = (const float*)d_in[3];
  const float* bq = (const float*)d_in[4];
  const float* Wk = (const float*)d_in[5];
  const float* bk = (const float*)d_in[6];
  const float* Wv = (const float*)d_in[7];
  const float* bv = (const float*)d_in[8];
  const float* Wp = (const float*)d_in[9];
  const float* bp = (const float*)d_in[10];

  float* out  = (float*)d_out;                       // [8,1024,1024]
  float* aavg = out + 8ull * 1024 * 1024;            // [8,1024,1024]

  const size_t SX = 8ull * 1024 * 1024;              // x / qkv elems
  const size_t SW = 1024ull * 1024;                  // weight elems
  __bf16* ws  = (__bf16*)d_ws;
  __bf16 *Xq = ws, *Xk = Xq + SX, *Xv = Xk + SX;
  __bf16 *Wqb = Xv + SX, *Wkb = Wqb + SW, *Wvb = Wkb + SW, *Wpb = Wvb + SW;
  __bf16 *Qb = Wpb + SW, *Kb = Qb + SX, *Vb = Kb + SX, *Ob = Vb + SX;

  const int NX = (int)SX, NW = (int)SW;
  xattn_cvt_f32_bf16<<<(NX + 255) / 256, 256, 0, stream>>>(xq, Xq, NX);
  xattn_cvt_f32_bf16<<<(NX + 255) / 256, 256, 0, stream>>>(xk, Xk, NX);
  xattn_cvt_f32_bf16<<<(NX + 255) / 256, 256, 0, stream>>>(xv, Xv, NX);
  xattn_cvt_f32_bf16<<<(NW + 255) / 256, 256, 0, stream>>>(Wq, Wqb, NW);
  xattn_cvt_f32_bf16<<<(NW + 255) / 256, 256, 0, stream>>>(Wk, Wkb, NW);
  xattn_cvt_f32_bf16<<<(NW + 255) / 256, 256, 0, stream>>>(Wv, Wvb, NW);
  xattn_cvt_f32_bf16<<<(NW + 255) / 256, 256, 0, stream>>>(Wp, Wpb, NW);
  xattn_zero_f32<<<(NX + 255) / 256, 256, 0, stream>>>(aavg, NX);

  dim3 gg(1024 / 128, 8192 / 128);   // (N tiles, M tiles)
  xattn_gemm_bf16_wmma<0><<<gg, 256, 0, stream>>>(Xq, Wqb, bq, Qb, nullptr);
  xattn_gemm_bf16_wmma<0><<<gg, 256, 0, stream>>>(Xk, Wkb, bk, Kb, nullptr);
  xattn_gemm_bf16_wmma<0><<<gg, 256, 0, stream>>>(Xv, Wvb, bv, Vb, nullptr);

  xattn_flash_wmma<<<dim3(128, 8), 256, 0, stream>>>(Qb, Kb, Vb, Ob, aavg);

  xattn_gemm_bf16_wmma<1><<<gg, 256, 0, stream>>>(Ob, Wpb, bp, nullptr, out);
}
